// IBQ_1159641170528
// MI455X (gfx1250) — matrix operand order repaired and checked by reading, not compiled, not run on hardware
//
#include <hip/hip_runtime.h>
#include <stdint.h>

typedef __attribute__((ext_vector_type(16))) _Float16 v16h;
typedef __attribute__((ext_vector_type(8)))  _Float16 v8h;
typedef __attribute__((ext_vector_type(8)))  float    v8f;
typedef __attribute__((ext_vector_type(4)))  float    v4f;

#define B_ROWS   32768
#define N_EMB    8192
#define E_DIM    256
#define MTILE    16
#define NTILE    16
#define KTILE    32
#define WAVES    8                     // waves per block
#define BLK      (WAVES * 32)          // 256 threads
#define MSUPER   (WAVES * MTILE)       // 128 rows of z per block
#define KTILES   (E_DIM / KTILE)       // 8
#define TILEHALF (NTILE * E_DIM)       // 4096 f16 = 8 KB per N-tile
#define NCHUNK   4                     // N-tiles per barrier
#define CHUNKHALF (NCHUNK * TILEHALF)  // 16384 f16 = 32 KB per chunk
#define NCHUNKS  (N_EMB / (NCHUNK * NTILE))   // 128

// ---------------------------------------------------------------------------
// Kernel 1: z (f32) -> z_h (f16). 8M elements, 4 per thread, exact grid.
// ---------------------------------------------------------------------------
__global__ void __launch_bounds__(BLK) vq_prep_z(const float* __restrict__ z,
                                                 _Float16* __restrict__ zh) {
    size_t i = ((size_t)blockIdx.x * BLK + threadIdx.x) * 4;
    v4f v = *(const v4f*)(z + i);
#pragma unroll
    for (int j = 0; j < 4; ++j) zh[i + j] = (_Float16)v[j];
}

// ---------------------------------------------------------------------------
// Kernel 2: embedding (f32) -> e_h (f16) + enorm[n] = sum_d e[n][d]^2 (f32).
// One wave per embedding row; lane handles 8 consecutive elements.
// ---------------------------------------------------------------------------
__global__ void __launch_bounds__(BLK) vq_prep_e(const float* __restrict__ e,
                                                 _Float16* __restrict__ eh,
                                                 float* __restrict__ enorm) {
    const int wave = threadIdx.x >> 5, lane = threadIdx.x & 31;
    const int row  = blockIdx.x * WAVES + wave;
    const float* src = e  + (size_t)row * E_DIM + lane * 8;
    _Float16*    dst = eh + (size_t)row * E_DIM + lane * 8;
    float s = 0.f;
#pragma unroll
    for (int j = 0; j < 8; ++j) { float v = src[j]; s += v * v; dst[j] = (_Float16)v; }
#pragma unroll
    for (int m = 16; m >= 1; m >>= 1) s += __shfl_xor(s, m, 32);
    if (lane == 0) enorm[row] = s;
}

// ---------------------------------------------------------------------------
// gfx1250 async copy: 32 KB contiguous global -> LDS, 128 B per thread
// (8x global_load_async_to_lds_b128, GVS mode: saddr SGPR pair + vgpr offset).
// Completion tracked with ASYNCcnt (s_wait_asynccnt), not LOADcnt.
// ---------------------------------------------------------------------------
__device__ __forceinline__ void async_chunk_copy(uint32_t lds_base,
                                                 const _Float16* gsrc, int tid) {
    uint64_t gp = (uint64_t)(uintptr_t)gsrc;
#pragma unroll
    for (int j = 0; j < 8; ++j) {
        uint32_t off = (uint32_t)tid * 128u + (uint32_t)j * 16u;
        uint32_t l   = lds_base + off;
        asm volatile("global_load_async_to_lds_b128 %0, %1, %2"
                     :: "v"(l), "v"(off), "s"(gp) : "memory");
    }
}

__device__ __forceinline__ void wait_async_all() {
    asm volatile("s_wait_asynccnt 0" ::: "memory");
}

// ---------------------------------------------------------------------------
// Kernel 3: WMMA GEMM + fused argmin.
//   score[b][n] = enorm[n] - 2 * dot(z_b, e_n)   (|z|^2 constant per row)
// Block = 8 waves, each wave owns a 16-row M-tile. Embedding is streamed as
// 32 KB chunks of 4 N-tiles, double-buffered in LDS via async-to-LDS and
// shared by all 8 waves (one asynccnt-wait + barrier per 64 columns).
// A fragments (z, f16, all K=256) live in 64 VGPRs/lane; 4 independent C
// accumulator chains are interleaved so consecutive WMMAs never RAW-chain.
// ---------------------------------------------------------------------------
__global__ void __launch_bounds__(BLK) vq_argmin(const _Float16* __restrict__ zh,
                                                 const _Float16* __restrict__ eh,
                                                 const float* __restrict__ enorm,
                                                 int* __restrict__ idx_out) {
    __shared__ __align__(16) _Float16 smem[2][CHUNKHALF];

    const int tid   = threadIdx.x;
    const int wave  = tid >> 5;
    const int lane  = tid & 31;
    const int g     = lane >> 4;   // lane group (K-half select)
    const int lm    = lane & 15;   // row (A) / column (B,C) within tile
    const int mbase = blockIdx.x * MSUPER + wave * MTILE;

    // ---- Preload A fragments: rows [mbase, mbase+16), all 8 K-tiles.
    // A layout (matches B): lanes 0-15 hold K 0-15, lanes 16-31 hold
    // K 16-31.
    v16h afrag[KTILES];
    const _Float16* zrow = zh + (size_t)(mbase + lm) * E_DIM;
#pragma unroll
    for (int kt = 0; kt < KTILES; ++kt) {
        const int k0 = kt * KTILE + g * 16;
        v8h lo = *(const v8h*)(zrow + k0);        // 16 B, aligned
        v8h hi = *(const v8h*)(zrow + k0 + 8);    // 16 B, aligned
        v16h a;
#pragma unroll
        for (int j = 0; j < 8; ++j) { a[j] = lo[j]; a[j + 8] = hi[j]; }
        afrag[kt] = a;
    }

    float mn[8];
    int   mi[8];
#pragma unroll
    for (int r = 0; r < 8; ++r) { mn[r] = 3.4e38f; mi[r] = 0; }

    const uint32_t lds0 = (uint32_t)(uintptr_t)(&smem[0][0]);
    const uint32_t lds1 = (uint32_t)(uintptr_t)(&smem[1][0]);

    // Prefetch chunk 0 into buffer 0.
    async_chunk_copy(lds0, eh, tid);
    wait_async_all();
    __syncthreads();

    for (int ch = 0; ch < NCHUNKS; ++ch) {
        const int cur = ch & 1;
        // Prefetch next chunk into the other buffer (safe: all waves passed
        // the barrier after finishing reads of that buffer last iteration).
        if (ch + 1 < NCHUNKS)
            async_chunk_copy(cur ? lds0 : lds1,
                             eh + (size_t)(ch + 1) * CHUNKHALF, tid);

        const _Float16* sb = &smem[cur][0];

        // ---- 4 x (16x16x256): 32 WMMAs, 4 independent accumulator chains.
        v8f c[NCHUNK];
#pragma unroll
        for (int t = 0; t < NCHUNK; ++t)
            c[t] = (v8f){0.f, 0.f, 0.f, 0.f, 0.f, 0.f, 0.f, 0.f};

#pragma unroll
        for (int kt = 0; kt < KTILES; ++kt) {
#pragma unroll
            for (int t = 0; t < NCHUNK; ++t) {
                // B layout (32x16 f16): lane = column n, lanes 0-15 K 0-15,
                // lanes 16-31 K 16-31 -> 16 contiguous f16 of embedding row.
                const v16h b = *(const v16h*)(sb + (size_t)t * TILEHALF
                                                 + (size_t)lm * E_DIM
                                                 + kt * KTILE + g * 16);
                c[t] = __builtin_amdgcn_wmma_f32_16x16x32_f16(
                           false, afrag[kt], false, b, (short)0, c[t],
                           false, false);
            }
        }

        // ---- Score + running argmin. C layout: VGPR r, lanes 0-15 -> row r,
        // lanes 16-31 -> row r+8; column = base + (lane&15) for all r.
#pragma unroll
        for (int t = 0; t < NCHUNK; ++t) {
            const int   ncol = (ch * NCHUNK + t) * NTILE + lm;
            const float en   = enorm[ncol];
#pragma unroll
            for (int r = 0; r < 8; ++r) {
                float sc = __builtin_fmaf(-2.f, c[t][r], en);
                if (sc < mn[r]) { mn[r] = sc; mi[r] = ncol; }
            }
        }

        wait_async_all();   // my async loads into next buffer complete
        __syncthreads();    // everyone's loads complete -> buffer ready
    }

    // ---- Cross-lane argmin within each 16-lane column group (wave32).
#pragma unroll
    for (int r = 0; r < 8; ++r) {
        float m = mn[r];
        int   i = mi[r];
#pragma unroll
        for (int s = 8; s >= 1; s >>= 1) {
            float om = __shfl_xor(m, s, 32);
            int   oi = __shfl_xor(i, s, 32);
            if (om < m || (om == m && oi < i)) { m = om; i = oi; }
        }
        if (lm == 0) idx_out[mbase + r + g * 8] = i;   // lanes 0 and 16 write
    }
}

// ---------------------------------------------------------------------------
// Kernel 4: z_q[b] = embedding[idx[b]] gather, float4 per thread.
// ---------------------------------------------------------------------------
__global__ void __launch_bounds__(BLK) vq_gather(const float* __restrict__ emb,
                                                 const int* __restrict__ idx,
                                                 float* __restrict__ zq) {
    size_t t   = (size_t)blockIdx.x * BLK + threadIdx.x;
    size_t row = t >> 6;          // 64 float4 per row
    size_t c4  = t & 63;
    int    n   = idx[row];
    ((v4f*)zq)[row * 64 + c4] = ((const v4f*)emb)[(size_t)n * 64 + c4];
}

// ---------------------------------------------------------------------------
extern "C" void kernel_launch(void* const* d_in, const int* in_sizes, int n_in,
                              void* d_out, int out_size, void* d_ws, size_t ws_size,
                              hipStream_t stream) {
    const float* z   = (const float*)d_in[0];  // [32768, 256] f32
    const float* emb = (const float*)d_in[1];  // [8192, 256]  f32

    float* zq  = (float*)d_out;                                  // output 0
    int*   idx = (int*)(zq + (size_t)B_ROWS * E_DIM);            // output 1 (int32)

    // Workspace layout: z_h (16 MB) | e_h (4 MB) | enorm (32 KB) ~= 20.2 MB
    _Float16* zh    = (_Float16*)d_ws;
    _Float16* ehalf = zh + (size_t)B_ROWS * E_DIM;
    float*    enorm = (float*)(ehalf + (size_t)N_EMB * E_DIM);

    vq_prep_z <<<(B_ROWS * E_DIM / 4) / BLK, BLK, 0, stream>>>(z, zh);
    vq_prep_e <<<N_EMB / WAVES,            BLK, 0, stream>>>(emb, ehalf, enorm);
    vq_argmin <<<B_ROWS / MSUPER,          BLK, 0, stream>>>(zh, ehalf, enorm, idx);
    vq_gather <<<(B_ROWS * 64) / BLK,      BLK, 0, stream>>>(emb, idx, zq);
}
